// DecoderWithAttention_33981781246337
// MI455X (gfx1250) — matrix operand order repaired and checked by reading, not compiled, not run on hardware
//
#include <hip/hip_runtime.h>
#include <hip/hip_bf16.h>

// ---------------- problem constants ----------------
#define B_    64
#define P_    196
#define ENC_  2048
#define A_    512
#define E_    512
#define D_    512
#define V_    20000
#define ML_   22
#define T_    21
#define XDIM_ (E_ + ENC_)   // 2560

typedef __attribute__((ext_vector_type(16))) __bf16 bfx16;
typedef __attribute__((ext_vector_type(8)))  __bf16 bfx8;
typedef __attribute__((ext_vector_type(8)))  float  fx8;

// ---------------------------------------------------------------------------
// Tiled transpose + f32 -> bf16 convert:  out[n*K + k] = (bf16) in[k*N + n]
// in is K x N row-major, out is N x K row-major. K,N multiples of 32.
// ---------------------------------------------------------------------------
__global__ __launch_bounds__(256)
void transpose_cvt_bf16(const float* __restrict__ in, __bf16* __restrict__ out,
                        int K, int N) {
  __shared__ float tile[32][33];
  const int n0 = blockIdx.x * 32;
  const int k0 = blockIdx.y * 32;
  const int tx = threadIdx.x;   // 0..31
  const int ty = threadIdx.y;   // 0..7
  #pragma unroll
  for (int i = 0; i < 32; i += 8)
    tile[ty + i][tx] = in[(size_t)(k0 + ty + i) * N + n0 + tx];
  __syncthreads();
  #pragma unroll
  for (int i = 0; i < 32; i += 8)
    out[(size_t)(n0 + ty + i) * K + k0 + tx] = (__bf16)tile[tx][ty + i];
}

// elementwise f32 -> bf16
__global__ void cvt_bf16_k(const float* __restrict__ in, __bf16* __restrict__ out, int n) {
  const int i = blockIdx.x * blockDim.x + threadIdx.x;
  if (i < n) out[i] = (__bf16)in[i];
}

// A-fragment (ISA 16-bit A 16x32 layout): elems 0..7 -> K=k+8*half+e,
// elems 8..15 -> K=k+16+8*half+(e-8). Two 16B loads, register concat.
__device__ __forceinline__ bfx16 frag_a(const __bf16* __restrict__ arow, int k, int half) {
  const bfx8 lo = *(const bfx8*)(arow + k + 8 * half);
  const bfx8 hi = *(const bfx8*)(arow + k + 16 + 8 * half);
  return __builtin_shufflevector(lo, hi, 0, 1, 2, 3, 4, 5, 6, 7,
                                 8, 9, 10, 11, 12, 13, 14, 15);
}

// ---------------------------------------------------------------------------
// WMMA bf16 GEMM (A already bf16):  C[M,N] = A[M,K] @ Bt[N,K]^T
// One wave -> one 16x16 tile; block = 8 waves = 8 N tiles; K step 64 (2 WMMA).
// lengths != nullptr => ragged-batch zero mask + non-temporal C store (fc path).
// ---------------------------------------------------------------------------
__global__ __launch_bounds__(256)
void gemm_wmma_bb(const __bf16* __restrict__ Abf, int lda,
                  const __bf16* __restrict__ Bt,
                  const float* __restrict__ bias0,
                  const float* __restrict__ bias1,
                  float* __restrict__ C, long long ldc,
                  int N, int K, int accum,
                  const int* __restrict__ lengths, int t) {
  const int lane = threadIdx.x & 31;
  const int wave = threadIdx.x >> 5;
  const int m0 = blockIdx.y << 4;
  const int n0 = (blockIdx.x << 7) + (wave << 4);
  if (n0 >= N) return;
  const int half = lane >> 4;
  const int l16  = lane & 15;
  const __bf16* arow = Abf + (size_t)(m0 + l16) * lda;
  const __bf16* brow = Bt  + (size_t)(n0 + l16) * K;

  fx8 acc = {0.f, 0.f, 0.f, 0.f, 0.f, 0.f, 0.f, 0.f};
  for (int k = 0; k < K; k += 64) {
    if (k + 64 < K) {
      __builtin_prefetch(brow + k + 64 + 16 * half, 0, 1);   // global_prefetch_b8
      __builtin_prefetch(arow + k + 64 + 8 * half, 0, 1);
    }
    const bfx16 av0 = frag_a(arow, k, half);
    const bfx16 bv0 = frag_a(brow, k, half);
    acc = __builtin_amdgcn_wmma_f32_16x16x32_bf16(false, av0, false, bv0,
                                                  (short)0, acc, false, false);
    const bfx16 av1 = frag_a(arow, k + 32, half);
    const bfx16 bv1 = frag_a(brow, k + 32, half);
    acc = __builtin_amdgcn_wmma_f32_16x16x32_bf16(false, av1, false, bv1,
                                                  (short)0, acc, false, false);
  }

  float bval = 0.f;
  if (!accum) {
    if (bias0) bval += bias0[n0 + l16];
    if (bias1) bval += bias1[n0 + l16];
  }
  #pragma unroll
  for (int j = 0; j < 8; ++j) {
    const int row = m0 + j + 8 * half;             // C/D layout: M = j (+8 upper half)
    const size_t idx = (size_t)row * (size_t)ldc + n0 + l16;
    float v = acc[j] + bval;
    if (accum) v += C[idx];
    if (lengths) {                                  // fc path: mask + NT store
      if (t >= lengths[row] - 1) v = 0.f;
      __builtin_nontemporal_store(v, &C[idx]);      // keep preds out of L2
    } else {
      C[idx] = v;
    }
  }
}

// ---------------------------------------------------------------------------
// WMMA bf16 GEMM with f32 A converted on the fly (used once, for att1 over
// encoder_out, to avoid a 50MB bf16 copy of the encoder in scratch).
// ---------------------------------------------------------------------------
__global__ __launch_bounds__(256)
void gemm_wmma_f32a(const float* __restrict__ Aa, int lda,
                    const __bf16* __restrict__ Bt,
                    const float* __restrict__ bias0,
                    float* __restrict__ C, long long ldc,
                    int N, int K) {
  const int lane = threadIdx.x & 31;
  const int wave = threadIdx.x >> 5;
  const int m0 = blockIdx.y << 4;
  const int n0 = (blockIdx.x << 7) + (wave << 4);
  if (n0 >= N) return;
  const int half = lane >> 4;
  const int l16  = lane & 15;
  const float*  arow = Aa + (size_t)(m0 + l16) * lda;
  const __bf16* brow = Bt + (size_t)(n0 + l16) * K;

  fx8 acc = {0.f, 0.f, 0.f, 0.f, 0.f, 0.f, 0.f, 0.f};
  for (int k = 0; k < K; k += 32) {
    if (k + 32 < K) {
      __builtin_prefetch(brow + k + 32 + 16 * half, 0, 1);
      __builtin_prefetch(arow + k + 32 + 8 * half, 0, 1);
    }
    const float4 a0 = *(const float4*)(arow + k + 8 * half);
    const float4 a1 = *(const float4*)(arow + k + 8 * half + 4);
    const float4 a2 = *(const float4*)(arow + k + 16 + 8 * half);
    const float4 a3 = *(const float4*)(arow + k + 16 + 8 * half + 4);
    bfx16 av;
    av[0]  = (__bf16)a0.x; av[1]  = (__bf16)a0.y; av[2]  = (__bf16)a0.z; av[3]  = (__bf16)a0.w;
    av[4]  = (__bf16)a1.x; av[5]  = (__bf16)a1.y; av[6]  = (__bf16)a1.z; av[7]  = (__bf16)a1.w;
    av[8]  = (__bf16)a2.x; av[9]  = (__bf16)a2.y; av[10] = (__bf16)a2.z; av[11] = (__bf16)a2.w;
    av[12] = (__bf16)a3.x; av[13] = (__bf16)a3.y; av[14] = (__bf16)a3.z; av[15] = (__bf16)a3.w;
    const bfx16 bv = frag_a(brow, k, half);
    acc = __builtin_amdgcn_wmma_f32_16x16x32_bf16(false, av, false, bv,
                                                  (short)0, acc, false, false);
  }
  const float bval = bias0 ? bias0[n0 + l16] : 0.f;
  #pragma unroll
  for (int j = 0; j < 8; ++j) {
    const int row = m0 + j + 8 * half;
    C[(size_t)row * (size_t)ldc + n0 + l16] = acc[j] + bval;
  }
}

// ---------------------------------------------------------------------------
// mean over pixels:  mean[b][e] = (1/P) * sum_p enc[b][p][e]
// ---------------------------------------------------------------------------
__global__ void mean_enc_k(const float* __restrict__ enc, float* __restrict__ mean) {
  const int idx = blockIdx.x * blockDim.x + threadIdx.x;
  if (idx >= B_ * ENC_) return;
  const int b = idx / ENC_, e = idx % ENC_;
  const float* base = enc + (size_t)b * P_ * ENC_ + e;
  float s = 0.f;
  for (int p = 0; p < P_; ++p) s += base[(size_t)p * ENC_];
  mean[idx] = s * (1.f / (float)P_);
}

// ---------------------------------------------------------------------------
// scores -> softmax -> context, fused per batch element (one block per b)
// ---------------------------------------------------------------------------
__global__ __launch_bounds__(256)
void attn_softmax_ctx(const float* __restrict__ att1,  // (B,P,A), pre-relu
                      const float* __restrict__ att2,  // (B,A)
                      const float* __restrict__ fw,    // (A)
                      const float* __restrict__ fb,    // (1)
                      const float* __restrict__ enc,   // (B,P,ENC)
                      float* __restrict__ context,     // (B,ENC)
                      float* __restrict__ alpha_out,   // alphas region base
                      const int* __restrict__ lengths, int t) {
  __shared__ float satt2[A_];
  __shared__ float swfa[A_];
  __shared__ float sp[P_];
  __shared__ float red[2];
  const int b = blockIdx.x, tid = threadIdx.x;

  for (int i = tid; i < A_; i += 256) { satt2[i] = att2[b * A_ + i]; swfa[i] = fw[i]; }
  __syncthreads();

  if (tid < P_) {
    const float* row = att1 + ((size_t)b * P_ + tid) * A_;
    float s = fb[0];
    for (int a = 0; a < A_; ++a) {
      const float v = row[a] + satt2[a];
      s += (v > 0.f ? v : 0.f) * swfa[a];
    }
    sp[tid] = s;
  }
  __syncthreads();
  if (tid == 0) {
    float mx = sp[0];
    for (int p = 1; p < P_; ++p) mx = fmaxf(mx, sp[p]);
    red[0] = mx;
  }
  __syncthreads();
  if (tid < P_) sp[tid] = __expf(sp[tid] - red[0]);
  __syncthreads();
  if (tid == 0) {
    float sm = 0.f;
    for (int p = 0; p < P_; ++p) sm += sp[p];
    red[1] = 1.f / sm;
  }
  __syncthreads();
  const bool active = t < lengths[b] - 1;
  if (tid < P_) {
    const float al = sp[tid] * red[1];
    sp[tid] = al;
    alpha_out[((size_t)b * T_ + t) * P_ + tid] = active ? al : 0.f;
  }
  __syncthreads();
  for (int e = tid; e < ENC_; e += 256) {           // coalesced in e across lanes
    const float* ebase = enc + (size_t)b * P_ * ENC_ + e;
    float acc = 0.f;
    for (int p = 0; p < P_; ++p) acc += sp[p] * ebase[(size_t)p * ENC_];
    context[b * ENC_ + e] = acc;
  }
}

// ---------------------------------------------------------------------------
// x(bf16) = concat(embedding lookup, sigmoid(gate) * context)
// ---------------------------------------------------------------------------
__global__ void build_x(const float* __restrict__ emb, const int* __restrict__ captions,
                        int t, const float* __restrict__ context,
                        const float* __restrict__ gatebuf, __bf16* __restrict__ x) {
  const int idx = blockIdx.x * blockDim.x + threadIdx.x;
  if (idx >= B_ * XDIM_) return;
  const int b = idx / XDIM_, j = idx % XDIM_;
  if (j < E_) {
    const int tok = captions[b * ML_ + t];
    x[idx] = (__bf16)emb[(size_t)tok * E_ + j];
  } else {
    const int e = j - E_;
    const float g = gatebuf[b * ENC_ + e];
    x[idx] = (__bf16)(context[b * ENC_ + e] * (1.f / (1.f + __expf(-g))));
  }
}

// ---------------------------------------------------------------------------
// LSTM pointwise (torch gate order i,f,g,o); c kept f32, h kept bf16
// ---------------------------------------------------------------------------
__global__ void lstm_cell(const float* __restrict__ gates, __bf16* __restrict__ h_bf,
                          float* __restrict__ c, __bf16* __restrict__ hnew_bf,
                          const int* __restrict__ lengths, int t) {
  const int idx = blockIdx.x * blockDim.x + threadIdx.x;
  if (idx >= B_ * D_) return;
  const int b = idx / D_, d = idx % D_;
  const float* g = gates + (size_t)b * 4 * D_;
  const float ig = 1.f / (1.f + __expf(-g[d]));
  const float fg = 1.f / (1.f + __expf(-g[D_ + d]));
  const float gg = tanhf(g[2 * D_ + d]);
  const float og = 1.f / (1.f + __expf(-g[3 * D_ + d]));
  const float cn = fg * c[idx] + ig * gg;
  const float hn = og * tanhf(cn);
  hnew_bf[idx] = (__bf16)hn;
  if (t < lengths[b] - 1) { h_bf[idx] = (__bf16)hn; c[idx] = cn; }
}

// ---------------------------------------------------------------------------
extern "C" void kernel_launch(void* const* d_in, const int* in_sizes, int n_in,
                              void* d_out, int out_size, void* d_ws, size_t ws_size,
                              hipStream_t stream) {
  (void)in_sizes; (void)n_in; (void)out_size; (void)ws_size;
  const float* encoder_out = (const float*)d_in[0];
  const int*   captions    = (const int*)  d_in[1];
  const int*   lengths     = (const int*)  d_in[2];
  const float* emb         = (const float*)d_in[3];
  const float* enc_att_w   = (const float*)d_in[4];
  const float* enc_att_b   = (const float*)d_in[5];
  const float* dec_att_w   = (const float*)d_in[6];
  const float* dec_att_b   = (const float*)d_in[7];
  const float* full_att_w  = (const float*)d_in[8];
  const float* full_att_b  = (const float*)d_in[9];
  const float* w_ih        = (const float*)d_in[10];
  const float* b_ih        = (const float*)d_in[11];
  const float* w_hh        = (const float*)d_in[12];
  const float* b_hh        = (const float*)d_in[13];
  const float* init_h_w    = (const float*)d_in[14];
  const float* init_h_b    = (const float*)d_in[15];
  const float* init_c_w    = (const float*)d_in[16];
  const float* init_c_b    = (const float*)d_in[17];
  const float* f_beta_w    = (const float*)d_in[18];
  const float* f_beta_b    = (const float*)d_in[19];
  const float* fc_w        = (const float*)d_in[20];
  const float* fc_b        = (const float*)d_in[21];

  float* out_preds  = (float*)d_out;                        // (B,T,V)
  float* out_alphas = (float*)d_out + (size_t)B_ * T_ * V_; // (B,T,P)

  // ---- workspace partition ----
  char* ws = (char*)d_ws;
  auto alloc_f = [&](size_t n) { float* p = (float*)ws;  ws += ((n * 4 + 255) / 256) * 256; return p; };
  auto alloc_b = [&](size_t n) { __bf16* p = (__bf16*)ws; ws += ((n * 2 + 255) / 256) * 256; return p; };

  float* meanb   = alloc_f((size_t)B_ * ENC_);
  float* c       = alloc_f((size_t)B_ * D_);
  float* htmp    = alloc_f((size_t)B_ * D_);
  float* att1    = alloc_f((size_t)B_ * P_ * A_);
  float* att2    = alloc_f((size_t)B_ * A_);
  float* context = alloc_f((size_t)B_ * ENC_);
  float* gatebuf = alloc_f((size_t)B_ * ENC_);
  float* gates   = alloc_f((size_t)B_ * 4 * D_);

  __bf16* mean_bf = alloc_b((size_t)B_ * ENC_);
  __bf16* h_bf    = alloc_b((size_t)B_ * D_);
  __bf16* hnew_bf = alloc_b((size_t)B_ * D_);
  __bf16* x_bf    = alloc_b((size_t)B_ * XDIM_);

  __bf16* enc_att_wT = alloc_b((size_t)A_ * ENC_);        // (A, ENC)
  __bf16* dec_att_wT = alloc_b((size_t)A_ * D_);          // (A, D)
  __bf16* init_h_wT  = alloc_b((size_t)D_ * ENC_);        // (D, ENC)
  __bf16* init_c_wT  = alloc_b((size_t)D_ * ENC_);
  __bf16* f_beta_wT  = alloc_b((size_t)ENC_ * D_);        // (ENC, D)
  __bf16* w_ihT      = alloc_b((size_t)(4 * D_) * XDIM_); // (4D, 2560)
  __bf16* w_hhT      = alloc_b((size_t)(4 * D_) * D_);    // (4D, D)
  __bf16* fc_wT      = alloc_b((size_t)V_ * D_);          // (V, D)

  // ---- weight transpose + bf16 convert (once per call) ----
  const dim3 tb(32, 8);
  auto xpose = [&](const float* in, __bf16* outp, int K, int N) {
    dim3 g(N / 32, K / 32);
    transpose_cvt_bf16<<<g, tb, 0, stream>>>(in, outp, K, N);
  };
  xpose(enc_att_w, enc_att_wT, ENC_, A_);
  xpose(dec_att_w, dec_att_wT, D_,   A_);
  xpose(init_h_w,  init_h_wT,  ENC_, D_);
  xpose(init_c_w,  init_c_wT,  ENC_, D_);
  xpose(f_beta_w,  f_beta_wT,  D_,   ENC_);
  xpose(w_ih,      w_ihT,      XDIM_, 4 * D_);
  xpose(w_hh,      w_hhT,      D_,   4 * D_);
  xpose(fc_w,      fc_wT,      D_,   V_);

  auto gemm = [&](const __bf16* Aa, int lda, const __bf16* Bt,
                  const float* bias0, const float* bias1,
                  float* Cc, long long ldc, int M, int N, int K,
                  int accum, const int* lens, int t) {
    dim3 g((N + 127) / 128, M / 16);
    gemm_wmma_bb<<<g, 256, 0, stream>>>(Aa, lda, Bt, bias0, bias1, Cc, ldc,
                                        N, K, accum, lens, t);
  };

  // ---- precompute ----
  mean_enc_k<<<(B_ * ENC_ + 255) / 256, 256, 0, stream>>>(encoder_out, meanb);
  cvt_bf16_k<<<(B_ * ENC_ + 255) / 256, 256, 0, stream>>>(meanb, mean_bf, B_ * ENC_);
  gemm(mean_bf, ENC_, init_h_wT, init_h_b, nullptr, htmp, D_, B_, D_, ENC_, 0, nullptr, 0);
  cvt_bf16_k<<<(B_ * D_ + 255) / 256, 256, 0, stream>>>(htmp, h_bf, B_ * D_);
  gemm(mean_bf, ENC_, init_c_wT, init_c_b, nullptr, c, D_, B_, D_, ENC_, 0, nullptr, 0);
  // att1 = encoder_out(flat) @ enc_att_w + b   -> (B*P, A); f32-A variant, runs once
  {
    dim3 g((A_ + 127) / 128, (B_ * P_) / 16);
    gemm_wmma_f32a<<<g, 256, 0, stream>>>(encoder_out, ENC_, enc_att_wT, enc_att_b,
                                          att1, A_, A_, ENC_);
  }

  // ---- decode steps ----
  for (int t = 0; t < T_; ++t) {
    gemm(h_bf, D_, dec_att_wT, dec_att_b, nullptr, att2, A_, B_, A_, D_, 0, nullptr, 0);
    attn_softmax_ctx<<<B_, 256, 0, stream>>>(att1, att2, full_att_w, full_att_b,
                                             encoder_out, context, out_alphas,
                                             lengths, t);
    gemm(h_bf, D_, f_beta_wT, f_beta_b, nullptr, gatebuf, ENC_, B_, ENC_, D_, 0, nullptr, 0);
    build_x<<<(B_ * XDIM_ + 255) / 256, 256, 0, stream>>>(emb, captions, t,
                                                          context, gatebuf, x_bf);
    gemm(x_bf, XDIM_, w_ihT, b_ih, b_hh, gates, 4 * D_, B_, 4 * D_, XDIM_, 0, nullptr, 0);
    gemm(h_bf, D_, w_hhT, nullptr, nullptr, gates, 4 * D_, B_, 4 * D_, D_, 1, nullptr, 0);
    lstm_cell<<<(B_ * D_ + 255) / 256, 256, 0, stream>>>(gates, h_bf, c, hnew_bf,
                                                         lengths, t);
    // preds[:, t, :] = mask(hnew @ fc_w + fc_b), NT-stored straight into d_out
    gemm(hnew_bf, D_, fc_wT, fc_b, nullptr, out_preds + (size_t)t * V_,
         (long long)T_ * V_, B_, V_, D_, 0, lengths, t);
  }
}
